// ImportanceRenderer_51608327029489
// MI455X (gfx1250) — matrix operand order repaired and checked by reading, not compiled, not run on hardware
//
#include <hip/hip_runtime.h>
#include <hip/hip_bf16.h>
#include <math.h>

#define NB 2
#define NR 4096
#define SC 32
#define SFI 32
#define FD 32
#define HID 64
#define OUTP 48   // 33 padded to 3x16

typedef _Float16 f16;
typedef __attribute__((ext_vector_type(16))) _Float16 v16h;
typedef __attribute__((ext_vector_type(8)))  _Float16 v8h;
typedef __attribute__((ext_vector_type(8)))  float    v8f;

typedef unsigned int u32x4 __attribute__((ext_vector_type(4)));
typedef int          i32x8 __attribute__((ext_vector_type(8)));
typedef int          i32x4 __attribute__((ext_vector_type(4)));

#if defined(__has_builtin)
#if __has_builtin(__builtin_amdgcn_tensor_load_to_lds)
#define HAVE_TDM 1
#endif
#endif

__device__ __forceinline__ float sigmoidf_(float x) { return 1.f / (1.f + __expf(-x)); }

// ---------------------------------------------------------------------------
// K0: global min of hit_min_depth, global max of hit_max_depth; init uint slots
// slots layout (16B): [0]=gmin f32, [1]=gmax f32, [2]=all-depth-min bits, [3]=all-depth-max bits
// ---------------------------------------------------------------------------
__launch_bounds__(256)
__global__ void minmax_hit_kernel(const float* __restrict__ hmn, const float* __restrict__ hmx,
                                  float* __restrict__ fslots, unsigned* __restrict__ uslots) {
    __shared__ float smn[256], smx[256];
    int tid = threadIdx.x;
    float mn = 1e30f, mx = -1e30f;
    for (int i = tid; i < NB * NR; i += 256) {
        mn = fminf(mn, hmn[i]);
        mx = fmaxf(mx, hmx[i]);
    }
    smn[tid] = mn; smx[tid] = mx;
    __syncthreads();
    for (int s = 128; s > 0; s >>= 1) {
        if (tid < s) {
            smn[tid] = fminf(smn[tid], smn[tid + s]);
            smx[tid] = fmaxf(smx[tid], smx[tid + s]);
        }
        __syncthreads();
    }
    if (tid == 0) {
        fslots[0] = smn[0];
        fslots[1] = smx[0];
        uslots[2] = 0x7F800000u;  // +inf bits (positive-float atomic min slot)
        uslots[3] = 0u;           // atomic max slot
    }
}

// ---------------------------------------------------------------------------
// K1: coarse depths  d = (base + jitter*span/32)*span + hmin
// ---------------------------------------------------------------------------
__launch_bounds__(256)
__global__ void coarse_depth_kernel(const float* __restrict__ hmn, const float* __restrict__ hmx,
                                    const float* __restrict__ jit, const int* __restrict__ mask,
                                    const float* __restrict__ fslots, float* __restrict__ dc) {
    int idx = blockIdx.x * 256 + threadIdx.x;
    if (idx >= NB * NR * SC) return;
    int s  = idx & 31;
    int rb = idx >> 5;
    int m  = mask[rb];
    float hmin = m ? hmn[rb] : fslots[0];
    float hmax = m ? hmx[rb] : fslots[1];
    float span = hmax - hmin;
    float base = (float)s * (1.f / 31.f);
    dc[idx] = (base + jit[idx] * span * (1.f / 32.f)) * span + hmin;
}

// ---------------------------------------------------------------------------
// K2/K6: tri-plane bilinear sampler. One wave per point, channel = lane (wave32).
// Writes f16 features row-major [point][32] (WMMA-A friendly) and tsdf.
// Planes (48MB) are L2-resident on the 192MB L2, so the scattered corner
// fetches are L2-bandwidth work, not HBM.
// ---------------------------------------------------------------------------
__launch_bounds__(256)
__global__ void sample_kernel(const float* __restrict__ planes, const float* __restrict__ ro,
                              const float* __restrict__ rdir, const float* __restrict__ depths,
                              f16* __restrict__ feat, float* __restrict__ tsdf) {
    int wv = threadIdx.x >> 5, ln = threadIdx.x & 31;
    int p  = blockIdx.x * 8 + wv;
    int rb = p >> 5;          // global ray
    int b  = rb / NR;
    float d = depths[p];
    const float* o3 = ro   + (size_t)rb * 3;
    const float* r3 = rdir + (size_t)rb * 3;
    float cx = o3[0] + d * r3[0];
    float cy = o3[1] + d * r3[1];
    float cz = o3[2] + d * r3[2];
    if (ln == 0) tsdf[p] = sqrtf(cx * cx + cy * cy + cz * cz) - 0.5f;
    float sx = cx * (1.f / 3.f), sy = cy * (1.f / 3.f), sz = cz * (1.f / 3.f);
    // plane projections: p0:(x,y)  p1:(x,z)  p2:(z,y)
    float px[3] = {sx, sx, sz};
    float py[3] = {sy, sz, sy};
    float acc = 0.f;
    for (int pl = 0; pl < 3; ++pl) {
        const float* fb = planes + (((size_t)(b * 3 + pl)) * FD + ln) * 65536;
        float x = (px[pl] + 1.f) * 128.f - 0.5f;
        float y = (py[pl] + 1.f) * 128.f - 0.5f;
        float x0f = floorf(x), y0f = floorf(y);
        float wx = x - x0f, wy = y - y0f;
        int x0 = (int)x0f, y0 = (int)y0f;
        int x0c = min(max(x0, 0), 255), y0c = min(max(y0, 0), 255);
        __builtin_prefetch(fb + y0c * 256 + x0c, 0, 0);  // global_prefetch_b8
        auto corner = [&](int yi, int xi) -> float {
            bool ok = (xi >= 0) && (xi < 256) && (yi >= 0) && (yi < 256);
            int xc = min(max(xi, 0), 255);
            int yc = min(max(yi, 0), 255);
            float v = fb[yc * 256 + xc];
            return ok ? v : 0.f;
        };
        acc += corner(y0, x0)         * (1.f - wx) * (1.f - wy)
             + corner(y0, x0 + 1)     * wx         * (1.f - wy)
             + corner(y0 + 1, x0)     * (1.f - wx) * wy
             + corner(y0 + 1, x0 + 1) * wx         * wy;
    }
    feat[(size_t)p * FD + ln] = (f16)(acc * (1.f / 3.f));
}

// ---------------------------------------------------------------------------
// K3/K7: WMMA MLP decoder.  16-point tiles per wave, 10 v_wmma per tile,
// 8 tiles per wave (256 blocks) so fragment setup amortizes.
//   layer1: feat[16x32] @ W1[32x64] + b1  -> softplus -> LDS (D->A relayout)
//   layer2: h[16x64]    @ W2[64x48] + b2  -> sigma/rgb/psdf
// W1 is DMA'd into LDS by the Tensor Data Mover when available.
// ---------------------------------------------------------------------------
__launch_bounds__(256)
__global__ void decoder_kernel(const f16* __restrict__ feat, const float* __restrict__ tsdf,
                               const int* __restrict__ mask,
                               const float* __restrict__ W1, const float* __restrict__ b1,
                               const float* __restrict__ W2, const float* __restrict__ b2,
                               const float* __restrict__ sbeta,
                               float* __restrict__ rgb_out, float* __restrict__ dens_out,
                               float* __restrict__ psdf_out, int pass_off) {
    __shared__ float sW1raw[32 * HID];      // 8KB f32 staging (TDM destination)
    __shared__ f16   sW1B[4 * 32 * 16];     // layer1 B fragments, [nt][lane][16]
    __shared__ f16   sW2B[6 * 32 * 16];     // layer2 B fragments, [nt*2+kt][lane][16]
    __shared__ float sB1[HID];
    __shared__ float sB2[OUTP];
    __shared__ f16   sH[8][16 * HID];       // per-wave hidden staging (2KB each)

    int t = threadIdx.x;

#ifdef HAVE_TDM
    if (t < 32) {
        // Tensor DMA descriptor (ISA 8.3/8.4): 2D tensor 64x32 f32, tile 64x32,
        // stride 64 elements, contiguous -> copies the whole 8KB W1 into LDS.
        unsigned long long ga = (unsigned long long)(uintptr_t)W1;
        unsigned lds_off = (unsigned)(uintptr_t)&sW1raw[0];
        u32x4 g0;
        g0[0] = 1u;                                             // count=1, user
        g0[1] = lds_off;                                        // lds_addr
        g0[2] = (unsigned)(ga & 0xFFFFFFFFu);                   // global_addr lo
        g0[3] = (unsigned)((ga >> 32) & 0x01FFFFFFu) | (2u << 30);  // addr hi | type=2
        i32x8 g1;
        g1[0] = (int)(2u << 16);      // data_size=2 (4B)
        g1[1] = (int)(64u << 16);     // tensor_dim0[15:0]=64  (bits 63:48)
        g1[2] = (int)(32u << 16);     // tensor_dim1[15:0]=32  (bits 95:80)
        g1[3] = (int)(64u << 16);     // tile_dim0=64          (bits 127:112)
        g1[4] = (int)32;              // tile_dim1=32, tile_dim2=0
        g1[5] = (int)64;              // tensor_dim0_stride=64
        g1[6] = 0;
        g1[7] = 0;
        i32x4 g2 = {0, 0, 0, 0};
        i32x4 g3 = {0, 0, 0, 0};
#if defined(__clang_major__) && (__clang_major__ >= 23)
        i32x8 g4 = {0, 0, 0, 0, 0, 0, 0, 0};
        __builtin_amdgcn_tensor_load_to_lds(g0, g1, g2, g3, g4, 0);
#else
        __builtin_amdgcn_tensor_load_to_lds(g0, g1, g2, g3, 0);
#endif
#if __has_builtin(__builtin_amdgcn_s_wait_tensorcnt)
        __builtin_amdgcn_s_wait_tensorcnt(0);
#else
        asm volatile("s_wait_tensorcnt 0x0" ::: "memory");
#endif
    }
#else
    for (int i = t; i < 32 * HID; i += 256) sW1raw[i] = W1[i];
#endif
    __syncthreads();

    // Pre-pack B fragments into LDS so each wave builds a fragment with one
    // contiguous 32B read (ds_load_b128 pair) instead of 16 strided gathers.
    for (int i = t; i < 4 * 32 * 16; i += 256) {
        int nt = i >> 9, r = i & 511, lnn = r >> 4, j = r & 15;
        int lrr = lnn & 15;
        int kbb = (lnn >= 16) ? 8 : 0;
        sW1B[i] = (f16)sW1raw[(kbb + (j & 7) + ((j & 8) << 1)) * HID + nt * 16 + lrr];
    }
    for (int i = t; i < 6 * 32 * 16; i += 256) {
        int f = i >> 9, r = i & 511, lnn = r >> 4, j = r & 15;
        int nt = f >> 1, kt = f & 1;
        int lrr = lnn & 15;
        int kbb = (lnn >= 16) ? 8 : 0;
        int n = nt * 16 + lrr;
        int K = kt * 32 + kbb + (j & 7) + ((j & 8) << 1);
        sW2B[i] = (f16)((n < 33) ? W2[K * 33 + n] : 0.f);
    }
    for (int i = t; i < HID;  i += 256) sB1[i] = b1[i];
    for (int i = t; i < OUTP; i += 256) sB2[i] = (i < 33) ? b2[i] : 0.f;
    __syncthreads();

    float beta = fmaxf(sbeta[0], 0.002f);
    float invb = 1.f / beta;
    int wv = t >> 5, ln = t & 31, lr = ln & 15;
    int o0 = (ln >= 16) ? 8 : 0;  // first K-octet for A frags
    int Mb = (ln >= 16) ? 8 : 0;  // M-base for C/D frags

    v16h B1f[4];
    #pragma unroll
    for (int nt = 0; nt < 4; ++nt)
        B1f[nt] = *(const v16h*)&sW1B[(nt * 32 + ln) * 16];
    v16h B2f[3][2];
    #pragma unroll
    for (int nt = 0; nt < 3; ++nt)
        #pragma unroll
        for (int kt = 0; kt < 2; ++kt)
            B2f[nt][kt] = *(const v16h*)&sW2B[((nt * 2 + kt) * 32 + ln) * 16];

    const int npoints = NB * NR * SC;
    const int tiles   = npoints / 16;
    f16* hs = &sH[wv][0];

    for (int tile = blockIdx.x * 8 + wv; tile < tiles; tile += gridDim.x * 8) {
        // A fragment: lane row = lr, K octets {o0, o0+16}
        const f16* fr = feat + (size_t)(tile * 16 + lr) * FD;
        union { v16h v; v8h h[2]; } au;
        au.h[0] = *(const v8h*)(fr + o0);
        au.h[1] = *(const v8h*)(fr + o0 + 16);
        v16h A = au.v;

        // ---- layer 1 ----
        #pragma unroll
        for (int nt = 0; nt < 4; ++nt) {
            float bias = sB1[nt * 16 + lr];
            v8f c;
            #pragma unroll
            for (int j = 0; j < 8; ++j) c[j] = bias;
            c = __builtin_amdgcn_wmma_f32_16x16x32_f16(false, A, false, B1f[nt],
                                                       (short)0, c, false, false);
            #pragma unroll
            for (int j = 0; j < 8; ++j) {
                float x = c[j];
                float h = fmaxf(x, 0.f) + __logf(1.f + __expf(-fabsf(x)));  // softplus
                hs[(Mb + j) * HID + nt * 16 + lr] = (f16)h;  // D-layout -> row major
            }
        }
        asm volatile("s_wait_dscnt 0x0" ::: "memory");  // same-wave LDS RAW

        // ---- layer 2 A fragments from LDS ----
        const f16* hr = hs + lr * HID;
        union { v16h v; v8h h[2]; } a2;
        a2.h[0] = *(const v8h*)(hr + o0);      a2.h[1] = *(const v8h*)(hr + o0 + 16);
        v16h A0 = a2.v;
        a2.h[0] = *(const v8h*)(hr + 32 + o0); a2.h[1] = *(const v8h*)(hr + 32 + o0 + 16);
        v16h A1 = a2.v;

        #pragma unroll
        for (int nt = 0; nt < 3; ++nt) {
            float bias = sB2[nt * 16 + lr];
            v8f c;
            #pragma unroll
            for (int j = 0; j < 8; ++j) c[j] = bias;
            c = __builtin_amdgcn_wmma_f32_16x16x32_f16(false, A0, false, B2f[nt][0],
                                                       (short)0, c, false, false);
            c = __builtin_amdgcn_wmma_f32_16x16x32_f16(false, A1, false, B2f[nt][1],
                                                       (short)0, c, false, false);
            int col = nt * 16 + lr;
            #pragma unroll
            for (int j = 0; j < 8; ++j) {
                int p = tile * 16 + Mb + j;
                float v = c[j];
                if (col == 0) {
                    int pb = p / (NR * SC);
                    int pi = p - pb * (NR * SC);
                    psdf_out[(size_t)pb * (2 * NR * SC) + pass_off + pi] = v;  // pre-mask
                    float mk = (float)mask[p >> 5];
                    float sg = v + tsdf[p];
                    dens_out[p] = sigmoidf_(-sg * invb) * invb * mk;
                } else if (col < 33) {
                    float mk = (float)mask[p >> 5];
                    rgb_out[(size_t)p * FD + (col - 1)] = (sigmoidf_(v) * 1.002f - 0.001f) * mk;
                }
            }
        }
    }
}

// ---------------------------------------------------------------------------
// K4: coarse ray march -> weights[31] per ray
// ---------------------------------------------------------------------------
__launch_bounds__(256)
__global__ void coarse_march_kernel(const float* __restrict__ dens, const float* __restrict__ dc,
                                    float* __restrict__ wout) {
    int ray = blockIdx.x * 256 + threadIdx.x;
    if (ray >= NB * NR) return;
    const float* dn = dens + (size_t)ray * 32;
    const float* dz = dc   + (size_t)ray * 32;
    float T = 1.f;
    for (int i = 0; i < 31; ++i) {
        float delta = dz[i + 1] - dz[i];
        float alpha = 1.f - __expf(-0.5f * (dn[i] + dn[i + 1]) * delta);
        float w = alpha * T;
        wout[(size_t)ray * 31 + i] = w;
        T *= (1.f - alpha + 1e-10f);
    }
}

// ---------------------------------------------------------------------------
// K5: importance sampling (max-filter smooth + inverse-CDF), thread per ray
// ---------------------------------------------------------------------------
__launch_bounds__(256)
__global__ void importance_kernel(const float* __restrict__ wc, const float* __restrict__ dc,
                                  const float* __restrict__ u, float* __restrict__ dfine) {
    int ray = blockIdx.x * 256 + threadIdx.x;
    if (ray >= NB * NR) return;
    float w[31], z[32];
    for (int i = 0; i < 31; ++i) w[i] = wc[(size_t)ray * 31 + i];
    for (int i = 0; i < 32; ++i) z[i] = dc[(size_t)ray * 32 + i];
    float zm[31];
    for (int i = 0; i < 31; ++i) zm[i] = 0.5f * (z[i] + z[i + 1]);
    float m[32];
    m[0] = fmaxf(-1e10f, w[0]);
    for (int i = 1; i < 31; ++i) m[i] = fmaxf(w[i - 1], w[i]);
    m[31] = fmaxf(w[30], -1e10f);
    float wsm[31];
    for (int i = 0; i < 31; ++i) wsm[i] = 0.5f * (m[i] + m[i + 1]) + 0.01f;
    // weights = wsm[1..29] (29 values), bins = zm (31 values)
    float pw[29]; float tot = 0.f;
    for (int i = 0; i < 29; ++i) { pw[i] = wsm[i + 1] + 1e-5f; tot += pw[i]; }
    float cdf[30]; cdf[0] = 0.f;
    for (int i = 0; i < 29; ++i) cdf[i + 1] = cdf[i] + pw[i] / tot;
    for (int k = 0; k < 32; ++k) {
        float uu = u[(size_t)ray * 32 + k];
        int ind = 0;
        for (int i = 0; i < 30; ++i) ind += (cdf[i] <= uu) ? 1 : 0;  // searchsorted right
        int below = (ind > 0) ? ind - 1 : 0;
        int above = (ind < 29) ? ind : 29;
        float c0 = cdf[below], c1 = cdf[above];
        float b0v = zm[below], b1v = zm[above];
        float den = c1 - c0;
        if (den < 1e-5f) den = 1.f;
        dfine[(size_t)ray * 32 + k] = b0v + (uu - c0) / den * (b1v - b0v);
    }
}

// ---------------------------------------------------------------------------
// K8: min/max over all depths (global), positive-float uint atomics
// ---------------------------------------------------------------------------
__launch_bounds__(256)
__global__ void depth_minmax_kernel(const float* __restrict__ dc, const float* __restrict__ df,
                                    unsigned* __restrict__ slots) {
    __shared__ float smn[256], smx[256];
    int tid = threadIdx.x;
    size_t i0 = (size_t)blockIdx.x * 256 + tid;
    size_t stride = (size_t)gridDim.x * 256;
    float mn = 1e30f, mx = -1e30f;
    const size_t n = (size_t)NB * NR * SC;
    for (size_t i = i0; i < n; i += stride) {
        float a = dc[i], b = df[i];
        mn = fminf(mn, fminf(a, b));
        mx = fmaxf(mx, fmaxf(a, b));
    }
    smn[tid] = mn; smx[tid] = mx;
    __syncthreads();
    for (int s = 128; s > 0; s >>= 1) {
        if (tid < s) {
            smn[tid] = fminf(smn[tid], smn[tid + s]);
            smx[tid] = fmaxf(smx[tid], smx[tid + s]);
        }
        __syncthreads();
    }
    if (tid == 0) {
        atomicMin(&slots[2], __float_as_uint(smn[0]));
        atomicMax(&slots[3], __float_as_uint(smx[0]));
    }
}

// ---------------------------------------------------------------------------
// K9: final merge+sort+march. Wave per ray, channel = lane. Stable rank sort
// of 64 depths in LDS (matches stable argsort of [coarse|fine]).
// ---------------------------------------------------------------------------
__launch_bounds__(256)
__global__ void final_march_kernel(const float* __restrict__ dc, const float* __restrict__ df,
                                   const float* __restrict__ rgbc, const float* __restrict__ rgbf,
                                   const float* __restrict__ densc, const float* __restrict__ densf,
                                   const unsigned* __restrict__ slots,
                                   float* __restrict__ rgb_out, float* __restrict__ depth_out,
                                   float* __restrict__ w_out) {
    __shared__ float sd[8][64];
    __shared__ int   si[8][64];
    int wv = threadIdx.x >> 5, ln = threadIdx.x & 31;
    int ray = blockIdx.x * 8 + wv;
    sd[wv][ln]      = dc[(size_t)ray * 32 + ln];
    sd[wv][ln + 32] = df[(size_t)ray * 32 + ln];
    asm volatile("s_wait_dscnt 0x0" ::: "memory");
    for (int e = ln; e < 64; e += 32) {
        float de = sd[wv][e];
        int rk = 0;
        for (int j = 0; j < 64; ++j) {
            float dj = sd[wv][j];
            rk += (dj < de) || (dj == de && j < e);  // stable
        }
        si[wv][rk] = e;
    }
    asm volatile("s_wait_dscnt 0x0" ::: "memory");

    auto loadC = [&](int i) -> float {
        const float* src = (i < 32) ? rgbc : rgbf;
        return src[((size_t)ray * 32 + (i & 31)) * FD + ln];
    };
    auto loadS = [&](int i) -> float {
        const float* src = (i < 32) ? densc : densf;
        return src[(size_t)ray * 32 + (i & 31)];
    };

    float T = 1.f, accC = 0.f, accD = 0.f, accW = 0.f;
    int   iP = si[wv][0];
    float zP = sd[wv][iP];
    float cP = loadC(iP), sP = loadS(iP);
    for (int k = 0; k < 63; ++k) {
        int   iN = si[wv][k + 1];
        float zN = sd[wv][iN];
        float cN = loadC(iN), sN = loadS(iN);
        float delta = zN - zP;
        float alpha = 1.f - __expf(-0.5f * (sP + sN) * delta);
        float w = alpha * T;
        accC += w * 0.5f * (cP + cN);
        accD += w * 0.5f * (zP + zN);
        accW += w;
        T *= (1.f - alpha + 1e-10f);
        if (ln == 0) w_out[(size_t)ray * 63 + k] = w;
        zP = zN; cP = cN; sP = sN;
    }
    rgb_out[(size_t)ray * 32 + ln] = accC * 2.f - 1.f;
    if (ln == 0) {
        float cdp = accD / accW;
        if (!(cdp == cdp)) cdp = INFINITY;  // nan -> inf
        float dmn = __uint_as_float(slots[2]);
        float dmx = __uint_as_float(slots[3]);
        depth_out[ray] = fminf(fmaxf(cdp, dmn), dmx);
    }
}

// ---------------------------------------------------------------------------
extern "C" void kernel_launch(void* const* d_in, const int* in_sizes, int n_in,
                              void* d_out, int out_size, void* d_ws, size_t ws_size,
                              hipStream_t stream) {
    const float* planes = (const float*)d_in[0];
    const float* ro     = (const float*)d_in[1];
    const float* rdir   = (const float*)d_in[2];
    const float* hmn    = (const float*)d_in[3];
    const float* hmx    = (const float*)d_in[4];
    const float* jit    = (const float*)d_in[5];
    const float* uimp   = (const float*)d_in[6];
    const float* W1     = (const float*)d_in[7];
    const float* b1     = (const float*)d_in[8];
    const float* W2     = (const float*)d_in[9];
    const float* b2     = (const float*)d_in[10];
    const float* sbeta  = (const float*)d_in[11];
    const int*   mask   = (const int*)d_in[12];

    char* ws = (char*)d_ws;
    const size_t P = (size_t)NB * NR * SC;  // 262144 points per pass
    float*    slots_f = (float*)ws;
    unsigned* slots_u = (unsigned*)ws;
    size_t off = 256;
    float* dc   = (float*)(ws + off); off += P * 4;
    float* df   = (float*)(ws + off); off += P * 4;
    f16*   feat = (f16*)  (ws + off); off += P * FD * 2;   // reused between passes
    float* tsdf = (float*)(ws + off); off += P * 4;        // reused between passes
    float* rgbc = (float*)(ws + off); off += P * FD * 4;
    float* rgbf = (float*)(ws + off); off += P * FD * 4;
    float* densc = (float*)(ws + off); off += P * 4;
    float* densf = (float*)(ws + off); off += P * 4;
    float* wc   = (float*)(ws + off); off += (size_t)NB * NR * 31 * 4;

    float* out       = (float*)d_out;
    float* out_rgb   = out;                               // NB*NR*32
    float* out_depth = out + NB * NR * FD;                // NB*NR
    float* out_w     = out_depth + NB * NR;               // NB*NR*63
    float* out_psdf  = out_w + (size_t)NB * NR * 63;      // NB*2*NR*SC

    minmax_hit_kernel<<<1, 256, 0, stream>>>(hmn, hmx, slots_f, slots_u);
    coarse_depth_kernel<<<(NB * NR * SC) / 256, 256, 0, stream>>>(hmn, hmx, jit, mask, slots_f, dc);
    sample_kernel<<<(int)(P / 8), 256, 0, stream>>>(planes, ro, rdir, dc, feat, tsdf);
    decoder_kernel<<<256, 256, 0, stream>>>(feat, tsdf, mask, W1, b1, W2, b2, sbeta,
                                            rgbc, densc, out_psdf, 0);
    coarse_march_kernel<<<(NB * NR) / 256, 256, 0, stream>>>(densc, dc, wc);
    importance_kernel<<<(NB * NR) / 256, 256, 0, stream>>>(wc, dc, uimp, df);
    sample_kernel<<<(int)(P / 8), 256, 0, stream>>>(planes, ro, rdir, df, feat, tsdf);
    decoder_kernel<<<256, 256, 0, stream>>>(feat, tsdf, mask, W1, b1, W2, b2, sbeta,
                                            rgbf, densf, out_psdf, NR * SC);
    depth_minmax_kernel<<<512, 256, 0, stream>>>(dc, df, slots_u);
    final_march_kernel<<<(NB * NR) / 8, 256, 0, stream>>>(dc, df, rgbc, rgbf, densc, densf,
                                                          slots_u, out_rgb, out_depth, out_w);
}
